// MoEAttentionPerExpert_88596585382522
// MI455X (gfx1250) — matrix operand order repaired and checked by reading, not compiled, not run on hardware
//
#include <hip/hip_runtime.h>
#include <hip/hip_bf16.h>
#include <math.h>
#include <stdint.h>

#define BATCH  16
#define SEQL   2048
#define DMODEL 512
#define NEXP   8
#define NHEAD  8
#define DHEAD  64
#define DFFN   2048
#define LN_EPSF 1e-5f
#define BETAF   0.5f
#define ALPHAF  1.0f

#define TTILE 64
#define TPAD  520                       // bf16 row stride for retrieved tile
#define TILE_BYTES (TTILE*TPAD*2)       // 66560
#define SSTR  2052                      // f32 per-head score stride (bank pad)
#define RSTR  520                       // f32 r_ctx row stride
#define NTILES (SEQL/TTILE)             // 32
#define WKE_SZ (32*16*24)               // swizzled wk_eff elems per (b,e) = 12288

typedef __attribute__((ext_vector_type(16))) __bf16 v16bf;
typedef __attribute__((ext_vector_type(8)))  float  v8f;

__device__ __forceinline__ v8f wmma_bf16(v16bf a, v16bf b, v8f c) {
  // D = A(16x32 bf16) * B(32x16 bf16) + C(16x16 f32)
  return __builtin_amdgcn_wmma_f32_16x16x32_bf16(false, a, false, b, (short)0, c, false, false);
}

// A-matrix (16-bit, 16x32): K index of packed element e (0..15) for a lane half.
__device__ __forceinline__ int a_kidx(int e, int half) {
  return (half << 4) + e;
}

// Issue one 64x512 bf16 tile as async global->LDS b128 copies (16 instr/wave).
__device__ __forceinline__ void issue_tile_async(const __bf16* g0, uint32_t lds0, int t) {
  for (int i = t; i < TTILE*DMODEL/8; i += 256) {
    int r = i >> 6, c8 = (i & 63) << 3;
    uint64_t g = (uint64_t)(uintptr_t)(g0 + (size_t)r*DMODEL + c8);
    uint32_t l = lds0 + (uint32_t)((r*TPAD + c8) << 1);
    asm volatile("global_load_async_to_lds_b128 %0, %1, off"
                 :: "v"(l), "v"(g) : "memory");
  }
}

// ---------------------------------------------------------------- f32 -> bf16
__global__ void k_cvt(const float* __restrict__ src, __bf16* __restrict__ dst, int n4) {
  int i = blockIdx.x*256 + threadIdx.x;
  if (i < n4) {
    float4 f = ((const float4*)src)[i];
    union { __bf16 h[4]; uint2 u; } o;
    o.h[0]=(__bf16)f.x; o.h[1]=(__bf16)f.y; o.h[2]=(__bf16)f.z; o.h[3]=(__bf16)f.w;
    ((uint2*)dst)[i] = o.u;
  }
}

// ---------------------------------------------------------------- gating
__global__ void k_gates(const float* __restrict__ query, const float* __restrict__ w_gate,
                        float* __restrict__ gates) {
  int b = blockIdx.x, t = threadIdx.x;
  __shared__ float lg[NEXP];
  if (t < NEXP) {
    float s = 0.f;
    for (int d = 0; d < DMODEL; ++d) s += query[b*DMODEL + d] * w_gate[d*NEXP + t];
    lg[t] = s;
  }
  __syncthreads();
  if (t < NEXP) {
    float m = lg[0];
    for (int e = 1; e < NEXP; ++e) m = fmaxf(m, lg[e]);
    float z = 0.f;
    for (int e = 0; e < NEXP; ++e) z += expf(lg[e] - m);
    gates[b*NEXP + t] = expf(lg[t] - m) / z;
  }
}

// ---------------- per-expert LN(q), qh, wk_eff (written WMMA-B swizzled)
__global__ void k_qproj(const float* __restrict__ query,
                        const float* __restrict__ ln_q_g, const float* __restrict__ ln_q_b,
                        const float* __restrict__ Wq, const float* __restrict__ bq,
                        const float* __restrict__ Wk,
                        __bf16* __restrict__ wkeff /* (B*E, WKE_SZ) swizzled */) {
  int be = blockIdx.x; int b = be / NEXP, e = be % NEXP;
  int t = threadIdx.x;
  __shared__ float qn[DMODEL];
  __shared__ float qh[DMODEL];
  __shared__ float red[256];

  float q0 = query[b*DMODEL + t], q1 = query[b*DMODEL + t + 256];
  red[t] = q0 + q1; __syncthreads();
  for (int o = 128; o > 0; o >>= 1) { if (t < o) red[t] += red[t+o]; __syncthreads(); }
  float mu = red[0] * (1.0f/DMODEL); __syncthreads();
  float d0 = q0 - mu, d1 = q1 - mu;
  red[t] = d0*d0 + d1*d1; __syncthreads();
  for (int o = 128; o > 0; o >>= 1) { if (t < o) red[t] += red[t+o]; __syncthreads(); }
  float rs = rsqrtf(red[0] * (1.0f/DMODEL) + LN_EPSF);
  __syncthreads();
  qn[t]     = d0*rs*ln_q_g[e*DMODEL+t]     + ln_q_b[e*DMODEL+t];
  qn[t+256] = d1*rs*ln_q_g[e*DMODEL+t+256] + ln_q_b[e*DMODEL+t+256];
  __syncthreads();

  for (int j = t; j < DMODEL; j += 256) {           // qh = qn @ Wq[e] + bq[e], *1/sqrt(HD)
    float acc = bq[e*DMODEL + j];
    const float* wcol = Wq + (size_t)e*DMODEL*DMODEL + j;
    for (int d = 0; d < DMODEL; ++d) acc += qn[d] * wcol[(size_t)d*DMODEL];
    qh[j] = acc * 0.125f;
  }
  __syncthreads();

  // wk_eff[K=d, n=h] stored as fragments: [((d>>4)*16 + h)*24 + (d&15)]
  __bf16* out = wkeff + (size_t)be*WKE_SZ;
  for (int d = t; d < DMODEL; d += 256) {
    const float* wrow = Wk + (size_t)e*DMODEL*DMODEL + (size_t)d*DMODEL;
    int grp = (d >> 4), el = d & 15;
    for (int h = 0; h < NHEAD; ++h) {
      float a = 0.f;
      for (int c = 0; c < DHEAD; ++c) a += wrow[h*DHEAD + c] * qh[h*DHEAD + c];
      out[(grp*16 + h)*24 + el] = (__bf16)a;
    }
    for (int h = NHEAD; h < 16; ++h) out[(grp*16 + h)*24 + el] = (__bf16)0.0f;
  }
}

// ------------------------------------- fused flash decode attention per (b,e)
__global__ void k_attn(const float* __restrict__ query,
                       const __bf16* __restrict__ rbf,
                       const __bf16* __restrict__ wkeff,
                       const float* __restrict__ Wv, const float* __restrict__ bv,
                       const float* __restrict__ Wo, const float* __restrict__ bo,
                       float* __restrict__ x_out, float* __restrict__ s_all) {
  extern __shared__ char smem[];
  // [ tile0 | tile1 | scores f32 8x2052 | wkS swz | pbuf swz | misc ]
  float*  scores = (float*)(smem + 2*TILE_BYTES);
  __bf16* wkS    = (__bf16*)((char*)scores + (size_t)NHEAD*SSTR*sizeof(float));
  __bf16* pbuf   = (__bf16*)((char*)wkS + (size_t)WKE_SZ*sizeof(__bf16));       // 64*2*8*16
  float*  misc   = (float*)((char*)pbuf + (size_t)(64*2*8*16)*sizeof(__bf16));  // 1024 f32
  float*  attnS  = misc;
  float*  mH     = misc + 512;
  float*  zH     = misc + 520;
  float*  red    = misc + 528;
  float*  rctx   = scores;  // reused after probs are consumed

  int be = blockIdx.x; int b = be / NEXP, e = be % NEXP;
  int t = threadIdx.x;
  int wave = t >> 5, lane = t & 31, half = lane >> 4, ln16 = lane & 15;
  uint32_t smem_base = (uint32_t)(uintptr_t)smem;

  const __bf16* wkg = wkeff + (size_t)be*WKE_SZ;
  for (int i = t; i < WKE_SZ; i += 256) wkS[i] = wkg[i];

  const __bf16* rbase = rbf + (size_t)b*SEQL*DMODEL;

  // ---- pass 1: scores[l,h] = tile @ wk_eff  (async double-buffered tiles)
  issue_tile_async(rbase, smem_base, t);
  for (int tt = 0; tt < NTILES; ++tt) {
    int cur = tt & 1;
    if (tt + 1 < NTILES) {
      issue_tile_async(rbase + (size_t)(tt+1)*TTILE*DMODEL,
                       smem_base + (uint32_t)(cur^1)*TILE_BYTES, t);
      asm volatile("s_wait_asynccnt 0x10" ::: "memory");   // tile tt complete
    } else {
      asm volatile("s_wait_asynccnt 0x0" ::: "memory");
    }
    __syncthreads();
    const __bf16* tile = (const __bf16*)(smem + (size_t)cur*TILE_BYTES);
    if (wave < 4) {                       // 4 M-tiles of 16 rows each
      v8f acc = {};
      for (int kc = 0; kc < DMODEL/32; ++kc) {
        const __bf16* pb = wkS + ((kc*2 + half)*16 + ln16)*24;
        v16bf a, bm;
        #pragma unroll
        for (int el = 0; el < 16; ++el) {
          a[el]  = tile[(wave*16 + ln16)*TPAD + kc*32 + a_kidx(el, half)];
          bm[el] = pb[el];
        }
        acc = wmma_bf16(a, bm, acc);
      }
      if (ln16 < NHEAD) {
        #pragma unroll
        for (int r = 0; r < 8; ++r) {
          int l = tt*TTILE + wave*16 + r + half*8;
          scores[ln16*SSTR + l] = acc[r];
        }
      }
    }
    __syncthreads();
  }

  // ---- per-head softmax (wave h owns head h)
  if (wave < NHEAD) {
    float m = -1e30f;
    for (int l = lane; l < SEQL; l += 32) m = fmaxf(m, scores[wave*SSTR + l]);
    for (int o = 16; o > 0; o >>= 1) m = fmaxf(m, __shfl_xor(m, o, 32));
    float z = 0.f;
    for (int l = lane; l < SEQL; l += 32) z += expf(scores[wave*SSTR + l] - m);
    for (int o = 16; o > 0; o >>= 1) z += __shfl_xor(z, o, 32);
    if (lane == 0) { mH[wave] = m; zH[wave] = 1.0f / z; }
  }
  __syncthreads();
  // normalize; also emit bf16 probs in WMMA-A swizzled layout (8 valid head rows)
  for (int i = t; i < NHEAD*SEQL; i += 256) {
    int h = i >> 11, l = i & (SEQL-1);
    float p = expf(scores[h*SSTR + l] - mH[h]) * zH[h];
    scores[h*SSTR + l] = p;
    int ahalf = (l >> 4) & 1, ael = l & 15, kc = l >> 5;
    pbuf[((kc*2 + ahalf)*8 + h)*16 + ael] = (__bf16)p;
  }
  __syncthreads();

  // ---- entropy of head-averaged probs -> reliability
  {
    float ent = 0.f;
    for (int l = t; l < SEQL; l += 256) {
      float pa = 0.f;
      for (int h = 0; h < NHEAD; ++h) pa += scores[h*SSTR + l];
      pa *= (1.0f/NHEAD);
      pa = fmaxf(pa, 1e-12f);
      ent -= pa * logf(pa);
    }
    red[t] = ent; __syncthreads();
    for (int o = 128; o > 0; o >>= 1) { if (t < o) red[t] += red[t+o]; __syncthreads(); }
    if (t == 0) s_all[be] = expf(-BETAF * red[0]);
    __syncthreads();
  }

  // ---- pass 2: r_ctx(16 x 512) = probs @ retrieved  (wave owns 64 columns)
  v8f acc[4] = {{},{},{},{}};
  issue_tile_async(rbase, smem_base, t);
  for (int tt = 0; tt < NTILES; ++tt) {
    int cur = tt & 1;
    if (tt + 1 < NTILES) {
      issue_tile_async(rbase + (size_t)(tt+1)*TTILE*DMODEL,
                       smem_base + (uint32_t)(cur^1)*TILE_BYTES, t);
      asm volatile("s_wait_asynccnt 0x10" ::: "memory");
    } else {
      asm volatile("s_wait_asynccnt 0x0" ::: "memory");
    }
    __syncthreads();
    const __bf16* tile = (const __bf16*)(smem + (size_t)cur*TILE_BYTES);
    for (int kc = 0; kc < 2; ++kc) {
      const __bf16* pa = pbuf + (((tt*2 + kc)*2 + half)*8 + (ln16 & 7))*16;
      v16bf a;
      #pragma unroll
      for (int el = 0; el < 16; ++el)
        a[el] = (ln16 < NHEAD) ? pa[el] : (__bf16)0.0f;
      #pragma unroll
      for (int nt = 0; nt < 4; ++nt) {
        v16bf bm;
        #pragma unroll
        for (int el = 0; el < 16; ++el)
          bm[el] = tile[(kc*32 + half*16 + el)*TPAD + wave*64 + nt*16 + ln16];
        acc[nt] = wmma_bf16(a, bm, acc[nt]);
      }
    }
    __syncthreads();
  }
  // spill r_ctx to LDS (overwrites scores region — all reads are done)
  #pragma unroll
  for (int nt = 0; nt < 4; ++nt) {
    #pragma unroll
    for (int r = 0; r < 8; ++r) {
      int h = r + half*8;
      if (h < NHEAD) rctx[h*RSTR + wave*64 + nt*16 + ln16] = acc[nt][r];
    }
  }
  __syncthreads();

  // ---- epilogue: attn = r_ctx @ Wv (per head) + bv ; x = query + attn @ Wo + bo
  for (int idx = t; idx < DMODEL; idx += 256) {
    int h = idx >> 6;
    const float* wv = Wv + (size_t)e*DMODEL*DMODEL + idx;   // column h*64+c == idx
    float a = bv[e*DMODEL + idx];
    for (int d = 0; d < DMODEL; ++d) a += rctx[h*RSTR + d] * wv[(size_t)d*DMODEL];
    attnS[idx] = a;
  }
  __syncthreads();
  for (int j = t; j < DMODEL; j += 256) {
    float s2 = bo[e*DMODEL + j];
    const float* wo = Wo + (size_t)e*DMODEL*DMODEL + j;
    for (int d = 0; d < DMODEL; ++d) s2 += attnS[d] * wo[(size_t)d*DMODEL];
    x_out[((size_t)b*NEXP + e)*DMODEL + j] = query[b*DMODEL + j] + s2;
  }
}

// ---------------------------------------------------------------- LN before FFN
__global__ void k_lno(const float* __restrict__ x, const float* __restrict__ g,
                      const float* __restrict__ bb, __bf16* __restrict__ xo) {
  int be = blockIdx.x; int b = be / NEXP, e = be % NEXP;
  int t = threadIdx.x;
  __shared__ float red[256];
  const float* xr = x + (size_t)be*DMODEL;
  float v0 = xr[t], v1 = xr[t+256];
  red[t] = v0 + v1; __syncthreads();
  for (int o = 128; o > 0; o >>= 1) { if (t < o) red[t] += red[t+o]; __syncthreads(); }
  float mu = red[0] * (1.0f/DMODEL); __syncthreads();
  float d0 = v0 - mu, d1 = v1 - mu;
  red[t] = d0*d0 + d1*d1; __syncthreads();
  for (int o = 128; o > 0; o >>= 1) { if (t < o) red[t] += red[t+o]; __syncthreads(); }
  float rs = rsqrtf(red[0] * (1.0f/DMODEL) + LN_EPSF);
  __bf16* out = xo + ((size_t)e*BATCH + b)*DMODEL;
  out[t]     = (__bf16)(d0*rs*g[e*DMODEL+t]     + bb[e*DMODEL+t]);
  out[t+256] = (__bf16)(d1*rs*g[e*DMODEL+t+256] + bb[e*DMODEL+t+256]);
}

// ------------------------------------------------- FFN1: h = gelu(xo @ W1 + b1)
__global__ void k_ffn1(const __bf16* __restrict__ xo, const float* __restrict__ W1,
                       const float* __restrict__ b1, __bf16* __restrict__ hbf) {
  int e = blockIdx.x >> 4, n0 = (blockIdx.x & 15) * 128;
  int t = threadIdx.x, wave = t >> 5, lane = t & 31, half = lane >> 4, ln16 = lane & 15;
  __shared__ __bf16 xoS[BATCH*DMODEL];
  __shared__ __bf16 wS[16*16*24];       // fragment-swizzled 32x128 weight tile
  for (int i = t; i < BATCH*DMODEL; i += 256) xoS[i] = xo[(size_t)e*BATCH*DMODEL + i];
  v8f acc = {};
  for (int kc = 0; kc < DMODEL/32; ++kc) {
    __syncthreads();
    for (int i = t; i < 32*128; i += 256) {
      int k = i >> 7, n = i & 127;
      float v = W1[(size_t)e*DMODEL*DFFN + (size_t)(kc*32 + k)*DFFN + n0 + n];
      wS[(((n>>4)*2 + (k>>4))*16 + (n&15))*24 + (k&15)] = (__bf16)v;
    }
    __syncthreads();
    const __bf16* pb = wS + ((wave*2 + half)*16 + ln16)*24;
    v16bf a, bm;
    #pragma unroll
    for (int el = 0; el < 16; ++el) {
      a[el]  = xoS[ln16*DMODEL + kc*32 + a_kidx(el, half)];
      bm[el] = pb[el];
    }
    acc = wmma_bf16(a, bm, acc);
  }
  #pragma unroll
  for (int r = 0; r < 8; ++r) {
    int brow = r + half*8;
    int col  = n0 + wave*16 + ln16;
    float v = acc[r] + b1[e*DFFN + col];
    v = 0.5f * v * (1.0f + erff(v * 0.70710678118f));     // exact GELU
    hbf[((size_t)e*BATCH + brow)*DFFN + col] = (__bf16)v;
  }
}

// -------------------------------------- FFN2: y = h @ W2 + b2 + x  (written to out)
__global__ void k_ffn2(const __bf16* __restrict__ hbf, const float* __restrict__ W2,
                       const float* __restrict__ b2, const float* __restrict__ x,
                       float* __restrict__ y_out) {
  int e = blockIdx.x >> 2, n0 = (blockIdx.x & 3) * 128;
  int t = threadIdx.x, wave = t >> 5, lane = t & 31, half = lane >> 4, ln16 = lane & 15;
  __shared__ __bf16 wS[16*16*24];
  v8f acc = {};
  for (int kc = 0; kc < DFFN/32; ++kc) {
    __syncthreads();
    for (int i = t; i < 32*128; i += 256) {
      int k = i >> 7, n = i & 127;
      float v = W2[(size_t)e*DFFN*DMODEL + (size_t)(kc*32 + k)*DMODEL + n0 + n];
      wS[(((n>>4)*2 + (k>>4))*16 + (n&15))*24 + (k&15)] = (__bf16)v;
    }
    __syncthreads();
    const __bf16* pb = wS + ((wave*2 + half)*16 + ln16)*24;
    v16bf a, bm;
    #pragma unroll
    for (int el = 0; el < 16; ++el) {
      a[el]  = hbf[((size_t)e*BATCH + ln16)*DFFN + kc*32 + a_kidx(el, half)];
      bm[el] = pb[el];
    }
    acc = wmma_bf16(a, bm, acc);
  }
  #pragma unroll
  for (int r = 0; r < 8; ++r) {
    int brow = r + half*8;
    int col  = n0 + wave*16 + ln16;
    float v = acc[r] + b2[e*DMODEL + col] + x[((size_t)brow*NEXP + e)*DMODEL + col];
    y_out[((size_t)brow*NEXP + e)*DMODEL + col] = v;
  }
}

// ---------------------------------------------------------------- final mixture
__global__ void k_combine(const float* __restrict__ gates, const float* __restrict__ s_all,
                          const float* __restrict__ y, const float* __restrict__ query,
                          float* __restrict__ fused, float* __restrict__ gt_out) {
  int b = blockIdx.x, t = threadIdx.x;
  __shared__ float gt[NEXP];
  __shared__ float inv;
  if (t < NEXP) gt[t] = gates[b*NEXP + t] * s_all[b*NEXP + t];
  __syncthreads();
  if (t == 0) {
    float s = 0.f;
    for (int e = 0; e < NEXP; ++e) s += gt[e];
    inv = 1.0f / (s + 1e-9f);
  }
  __syncthreads();
  if (t < NEXP) gt[t] *= inv;
  __syncthreads();
  for (int d = t; d < DMODEL; d += 256) {
    float mix = 0.f;
    for (int e = 0; e < NEXP; ++e) mix += gt[e] * y[((size_t)b*NEXP + e)*DMODEL + d];
    fused[b*DMODEL + d] = ALPHAF*mix + (1.0f - ALPHAF)*query[b*DMODEL + d];
  }
  if (t < NEXP) gt_out[b*NEXP + t] = gt[t];
}

// ---------------------------------------------------------------- launch
extern "C" void kernel_launch(void* const* d_in, const int* in_sizes, int n_in,
                              void* d_out, int out_size, void* d_ws, size_t ws_size,
                              hipStream_t stream) {
  const float* query     = (const float*)d_in[0];
  const float* retrieved = (const float*)d_in[1];
  const float* w_gate    = (const float*)d_in[2];
  const float* ln_q_g    = (const float*)d_in[3];
  const float* ln_q_b    = (const float*)d_in[4];
  const float* Wq        = (const float*)d_in[5];
  const float* bq        = (const float*)d_in[6];
  const float* Wk        = (const float*)d_in[7];
  // d_in[8] = bk: constant-per-head score shift -> softmax-invariant, exactly droppable
  const float* Wv        = (const float*)d_in[9];
  const float* bv        = (const float*)d_in[10];
  const float* Wo        = (const float*)d_in[11];
  const float* bo        = (const float*)d_in[12];
  const float* ln_o_g    = (const float*)d_in[13];
  const float* ln_o_b    = (const float*)d_in[14];
  const float* W1        = (const float*)d_in[15];
  const float* b1        = (const float*)d_in[16];
  const float* W2        = (const float*)d_in[17];
  const float* b2        = (const float*)d_in[18];

  char* ws = (char*)d_ws;
  float*  gates = (float*)(ws + 0);                         // 128 f32
  float*  sall  = (float*)(ws + 1024);                      // 128 f32
  __bf16* wkeff = (__bf16*)(ws + 2048);                     // 128*12288*2 = 3,145,728
  float*  x     = (float*)(ws + 2048 + 3145728);            // 256 KB
  __bf16* xo    = (__bf16*)(ws + 2048 + 3145728 + 262144);  // 128 KB
  __bf16* hbf   = (__bf16*)(ws + 2048 + 3145728 + 262144 + 131072);            // 512 KB
  __bf16* rbf   = (__bf16*)(ws + 2048 + 3145728 + 262144 + 131072 + 524288);   // 32 MB

  float* fused = (float*)d_out;                    // (B, D)
  float* y     = fused + BATCH*DMODEL;             // (B, E, D)
  float* gt    = y + BATCH*NEXP*DMODEL;            // (B, E)

  k_cvt<<<(BATCH*SEQL*DMODEL/4 + 255)/256, 256, 0, stream>>>(retrieved, rbf,
                                                             BATCH*SEQL*DMODEL/4);
  k_gates<<<BATCH, 64, 0, stream>>>(query, w_gate, gates);
  k_qproj<<<BATCH*NEXP, 256, 0, stream>>>(query, ln_q_g, ln_q_b, Wq, bq, Wk, wkeff);

  size_t attn_lds = 2*(size_t)TILE_BYTES                    // double-buffered tiles
                  + (size_t)NHEAD*SSTR*sizeof(float)        // scores
                  + (size_t)WKE_SZ*sizeof(__bf16)           // swizzled wk_eff
                  + (size_t)(64*2*8*16)*sizeof(__bf16)      // swizzled probs
                  + 1024*sizeof(float);                     // misc  (~254 KB / 320 KB)
  k_attn<<<BATCH*NEXP, 256, attn_lds, stream>>>(query, rbf, wkeff, Wv, bv, Wo, bo, x, sall);

  k_lno<<<BATCH*NEXP, 256, 0, stream>>>(x, ln_o_g, ln_o_b, xo);
  k_ffn1<<<NEXP*16, 256, 0, stream>>>(xo, W1, b1, hbf);
  k_ffn2<<<NEXP*4,  256, 0, stream>>>(hbf, W2, b2, x, y);
  k_combine<<<BATCH, 256, 0, stream>>>(gates, sall, y, query, fused, gt);

  (void)in_sizes; (void)n_in; (void)out_size; (void)ws_size;
}
